// Faster_MultiHeadAttention_38010460570337
// MI455X (gfx1250) — matrix operand order repaired and checked by reading, not compiled, not run on hardware
//
#include <hip/hip_runtime.h>

#define HID 512
#define NH  8
#define VD  64
#define NB  2
#define SEQ 2048
#define SCALE2 0.1803368801111244f   // (1/sqrt(64)) * log2(e)

typedef __attribute__((ext_vector_type(16))) _Float16 v16h;
typedef __attribute__((ext_vector_type(8)))  _Float16 v8h;
typedef __attribute__((ext_vector_type(8)))  float    v8f;
typedef __attribute__((ext_vector_type(4)))  int      v4i;

// ---- async global->LDS copy (16 bytes per lane), with safe fallback ----
#if __has_builtin(__builtin_amdgcn_global_load_async_to_lds_b128)
#define ASYNC_CP16(l, g)                                                      \
  __builtin_amdgcn_global_load_async_to_lds_b128(                             \
      (__attribute__((address_space(1))) v4i*)(unsigned long long)(const void*)(g), \
      (__attribute__((address_space(3))) v4i*)(unsigned int)(unsigned long long)(void*)(l), \
      0, 0)
#define ASYNC_WAIT() asm volatile("s_wait_asynccnt 0x0" ::: "memory")
#else
#define ASYNC_CP16(l, g) (*(v8h*)(l) = *(const v8h*)(g))
#define ASYNC_WAIT() ((void)0)
#endif

__device__ __forceinline__ v8f wmma_f16(v16h a, v16h b, v8f c) {
  return __builtin_amdgcn_wmma_f32_16x16x32_f16(
      false, a, false, b, (short)0, c, false, false);
}

// A fragment: 16x32 tile of a row-major [16 x ld] f16 matrix at p, columns k0..k0+31.
__device__ __forceinline__ v16h frag_a(const _Float16* p, int ld, int k0) {
  int lane = threadIdx.x & 31;
  int r = lane & 15, ks = lane >> 4;
  const _Float16* b0 = p + r * ld + k0 + ks * 8;
  v8h lo = *(const v8h*)b0;          // K = k0 + ks*8 .. +7
  v8h hi = *(const v8h*)(b0 + 16);   // K = k0 + 16 + ks*8 .. +7
  v16h f;
#pragma unroll
  for (int i = 0; i < 8; ++i) { f[i] = lo[i]; f[i + 8] = hi[i]; }
  return f;
}

// B fragment: 32x16 tile. T is column-contiguous storage: T[n][k] row-major, ld.
__device__ __forceinline__ v16h frag_b(const _Float16* t, int ld, int k0) {
  int lane = threadIdx.x & 31;
  int n = lane & 15, kh = lane >> 4;
  const _Float16* b0 = t + n * ld + k0 + kh * 8;
  v8h lo = *(const v8h*)b0;
  v8h hi = *(const v8h*)(b0 + 16);
  v16h f;
#pragma unroll
  for (int i = 0; i < 8; ++i) { f[i] = lo[i]; f[i + 8] = hi[i]; }
  return f;
}

__device__ __forceinline__ float redmax16(float v) {
  v = fmaxf(v, __shfl_xor(v, 1, 16));
  v = fmaxf(v, __shfl_xor(v, 2, 16));
  v = fmaxf(v, __shfl_xor(v, 4, 16));
  v = fmaxf(v, __shfl_xor(v, 8, 16));
  return v;
}
__device__ __forceinline__ float redsum16(float v) {
  v += __shfl_xor(v, 1, 16);
  v += __shfl_xor(v, 2, 16);
  v += __shfl_xor(v, 4, 16);
  v += __shfl_xor(v, 8, 16);
  return v;
}

// ---------------- conversion / packing ----------------

__global__ __launch_bounds__(256) void cvt_x_kernel(const float* __restrict__ X,
                                                    _Float16* __restrict__ Xh, int n) {
  int i = blockIdx.x * blockDim.x + threadIdx.x;
  if (i < n) Xh[i] = (_Float16)X[i];
}

// WqT[n][d] = Wq[h][d][e] (n=h*64+e);  WoT[o][d] = Wo[d][o]
__global__ __launch_bounds__(256) void pack_w_kernel(
    const float* __restrict__ Wq, const float* __restrict__ Wk,
    const float* __restrict__ Wv, const float* __restrict__ Wo,
    _Float16* __restrict__ WqT, _Float16* __restrict__ WkT,
    _Float16* __restrict__ WvT, _Float16* __restrict__ WoT) {
  int i = blockIdx.x * blockDim.x + threadIdx.x;   // over 512*512
  int n = i >> 9, d = i & 511;
  int h = n >> 6, e = n & 63;
  WqT[i] = (_Float16)Wq[(h * HID + d) * VD + e];
  WkT[i] = (_Float16)Wk[(h * HID + d) * VD + e];
  WvT[i] = (_Float16)Wv[(h * HID + d) * VD + e];
  WoT[i] = (_Float16)Wo[d * HID + n];
}

// ---------------- QKV projection (register double-buffered) ----------------
__global__ __launch_bounds__(256) void qkv_kernel(
    const _Float16* __restrict__ Xh,
    const _Float16* __restrict__ WqT, const _Float16* __restrict__ WkT,
    const _Float16* __restrict__ WvT,
    const float* __restrict__ bq, const float* __restrict__ bk,
    const float* __restrict__ bv,
    _Float16* __restrict__ Qh, _Float16* __restrict__ Kh,
    _Float16* __restrict__ Vt) {
  int wid = blockIdx.x * (blockDim.x >> 5) + (threadIdx.x >> 5);
  int mt = wid >> 5;       // 0..255 row tile over B*S
  int nt = wid & 31;       // 0..31  col tile over H*VD
  int lane = threadIdx.x & 31;

  const _Float16* abase  = Xh  + (size_t)(mt * 16) * HID;
  const _Float16* wqBase = WqT + (size_t)(nt * 16) * HID;
  const _Float16* wkBase = WkT + (size_t)(nt * 16) * HID;
  const _Float16* wvBase = WvT + (size_t)(nt * 16) * HID;

  v8f cq = {}, ck = {}, cv = {};
  v16h a  = frag_a(abase, HID, 0);
  v16h fq = frag_b(wqBase, HID, 0);
  v16h fk = frag_b(wkBase, HID, 0);
  v16h fv = frag_b(wvBase, HID, 0);
#pragma unroll 2
  for (int kk = 0; kk < HID - 32; kk += 32) {
    v16h a2  = frag_a(abase, HID, kk + 32);   // prefetch next chunk
    v16h fq2 = frag_b(wqBase, HID, kk + 32);
    v16h fk2 = frag_b(wkBase, HID, kk + 32);
    v16h fv2 = frag_b(wvBase, HID, kk + 32);
    cq = wmma_f16(a, fq, cq);
    ck = wmma_f16(a, fk, ck);
    cv = wmma_f16(a, fv, cv);
    a = a2; fq = fq2; fk = fk2; fv = fv2;
  }
  cq = wmma_f16(a, fq, cq);
  ck = wmma_f16(a, fk, ck);
  cv = wmma_f16(a, fv, cv);

  int n = lane & 15, kh = lane >> 4;
  int col = nt * 16 + n;
  int h = col >> 6, e = col & 63;
  float biasq = bq[col], biask = bk[col], biasv = bv[col];

  int b  = mt >> 7;
  int s0 = (mt & 127) * 16;
  _Float16* qp = Qh + (size_t)((b * NH + h) * SEQ + s0) * VD + e;
  _Float16* kp = Kh + (size_t)((b * NH + h) * SEQ + s0) * VD + e;

  v8h vv;
#pragma unroll
  for (int i = 0; i < 8; ++i) {
    int m = i + 8 * kh;
    qp[(size_t)m * VD] = (_Float16)(cq[i] + biasq);
    kp[(size_t)m * VD] = (_Float16)(ck[i] + biask);
    vv[i] = (_Float16)(cv[i] + biasv);
  }
  _Float16* vp = Vt + (size_t)((b * NH + h) * VD + e) * SEQ + s0 + 8 * kh;
  *(v8h*)vp = vv;   // transposed V store, one b128
}

// ---------------- flash attention (block-shared K/V tiles, async copy) ----------------
// Block = 8 waves, all on the same (b,h); wave w handles query tile st_base + w.
__global__ __launch_bounds__(256) void attn_kernel(
    const _Float16* __restrict__ Qh, const _Float16* __restrict__ Kh,
    const _Float16* __restrict__ Vt, _Float16* __restrict__ Zh) {
  __shared__ _Float16 Ktile[2][32 * 64];   // [t_local][e]
  __shared__ _Float16 Vtile[2][64 * 32];   // [e][t_local]
  __shared__ _Float16 Ptile[8][16 * 32];   // per-wave P staging

  int tid = threadIdx.x;
  int wslot = tid >> 5, lane = tid & 31;
  int bh = blockIdx.x >> 4;                 // 16 blocks per (b,h)
  int st = (blockIdx.x & 15) * 8 + wslot;   // query tile 0..127

  const _Float16* qbase = Qh + (size_t)(bh * SEQ + st * 16) * VD;
  const _Float16* kbase = Kh + (size_t)bh * SEQ * VD;
  const _Float16* vtb   = Vt + (size_t)bh * VD * SEQ;
  _Float16* pw = &Ptile[wslot][0];

  // cooperative stage of one 32-key step into LDS buffer `buf`
  auto stage = [&](int buf, int t0) {
    int krow = tid >> 3, kseg = tid & 7;   // 32 rows x 128B
    ASYNC_CP16(&Ktile[buf][krow * 64 + kseg * 8],
               kbase + (size_t)(t0 + krow) * VD + kseg * 8);
    int e = tid >> 2, vseg = tid & 3;      // 64 rows x 64B
    ASYNC_CP16(&Vtile[buf][e * 32 + vseg * 8],
               vtb + (size_t)e * SEQ + t0 + vseg * 8);
  };

  v16h qa0 = frag_a(qbase, VD, 0);
  v16h qa1 = frag_a(qbase, VD, 32);

  v8f acc0 = {}, acc1 = {}, acc2 = {}, acc3 = {};
  float rmax[8], rsum[8], fac[8];
#pragma unroll
  for (int i = 0; i < 8; ++i) { rmax[i] = -3.0e38f; rsum[i] = 0.0f; }

  int n = lane & 15, kh = lane >> 4;

  stage(0, 0);
  for (int t0 = 0; t0 < SEQ; t0 += 32) {
    int buf = (t0 >> 5) & 1;
    ASYNC_WAIT();
    __syncthreads();                       // buf ready; buf^1 free to refill
    if (t0 + 32 < SEQ) stage(buf ^ 1, t0 + 32);

    const _Float16* kt = &Ktile[buf][0];
    const _Float16* vt = &Vtile[buf][0];

    // scores: 16 queries x 32 keys, contraction over VD=64
    v8f s0 = {}, s1 = {};
    s0 = wmma_f16(qa0, frag_b(kt, 64, 0),  s0);
    s0 = wmma_f16(qa1, frag_b(kt, 64, 32), s0);
    s1 = wmma_f16(qa0, frag_b(kt + 16 * 64, 64, 0),  s1);
    s1 = wmma_f16(qa1, frag_b(kt + 16 * 64, 64, 32), s1);

    // online softmax in base-2 domain (v_exp_f32 is native exp2)
#pragma unroll
    for (int i = 0; i < 8; ++i) {
      float a0 = s0[i] * SCALE2;
      float a1 = s1[i] * SCALE2;
      float mx = redmax16(fmaxf(a0, a1));
      float nm = fmaxf(rmax[i], mx);
      float f  = exp2f(rmax[i] - nm);
      rmax[i] = nm;
      float p0 = exp2f(a0 - nm);
      float p1 = exp2f(a1 - nm);
      rsum[i] = rsum[i] * f + redsum16(p0 + p1);
      fac[i] = f;
      int m = i + 8 * kh;
      pw[m * 32 + n]      = (_Float16)p0;
      pw[m * 32 + 16 + n] = (_Float16)p1;
    }
#pragma unroll
    for (int i = 0; i < 8; ++i) {
      acc0[i] *= fac[i]; acc1[i] *= fac[i];
      acc2[i] *= fac[i]; acc3[i] *= fac[i];
    }

    // wave-local LDS transpose of P (DS ops are wave-ordered)
    asm volatile("s_wait_dscnt 0x0" ::: "memory");
    v16h pa = frag_a(pw, 32, 0);

    acc0 = wmma_f16(pa, frag_b(vt + 0 * 16 * 32, 32, 0), acc0);
    acc1 = wmma_f16(pa, frag_b(vt + 1 * 16 * 32, 32, 0), acc1);
    acc2 = wmma_f16(pa, frag_b(vt + 2 * 16 * 32, 32, 0), acc2);
    acc3 = wmma_f16(pa, frag_b(vt + 3 * 16 * 32, 32, 0), acc3);
  }

  int b = bh >> 3, h = bh & 7;
  v8f accs[4] = {acc0, acc1, acc2, acc3};
#pragma unroll
  for (int es = 0; es < 4; ++es) {
#pragma unroll
    for (int i = 0; i < 8; ++i) {
      int m = i + 8 * kh;
      int s = st * 16 + m;
      int e = es * 16 + n;
      Zh[(size_t)(b * SEQ + s) * HID + h * VD + e] =
          (_Float16)(accs[es][i] / rsum[i]);
    }
  }
}

// ---------------- output projection: 16x32 tile per wave, pipelined ----------------
__global__ __launch_bounds__(256) void proj_kernel(
    const _Float16* __restrict__ Zh, const _Float16* __restrict__ WoT,
    const float* __restrict__ bo, float* __restrict__ out) {
  int wid = blockIdx.x * (blockDim.x >> 5) + (threadIdx.x >> 5);
  int mt = wid >> 4;        // 0..255
  int np = wid & 15;        // 16 column-pairs of 32
  int lane = threadIdx.x & 31;

  const _Float16* abase = Zh  + (size_t)(mt * 16) * HID;
  const _Float16* w0    = WoT + (size_t)(np * 32) * HID;
  const _Float16* w1    = w0 + (size_t)16 * HID;

  v8f c0 = {}, c1 = {};
  v16h a  = frag_a(abase, HID, 0);
  v16h f0 = frag_b(w0, HID, 0);
  v16h f1 = frag_b(w1, HID, 0);
#pragma unroll 2
  for (int kk = 0; kk < HID - 32; kk += 32) {
    v16h a2  = frag_a(abase, HID, kk + 32);
    v16h f02 = frag_b(w0, HID, kk + 32);
    v16h f12 = frag_b(w1, HID, kk + 32);
    c0 = wmma_f16(a, f0, c0);
    c1 = wmma_f16(a, f1, c1);
    a = a2; f0 = f02; f1 = f12;
  }
  c0 = wmma_f16(a, f0, c0);
  c1 = wmma_f16(a, f1, c1);

  int n = lane & 15, kh = lane >> 4;
  int col0 = np * 32 + n;
  float b0 = bo[col0], b1 = bo[col0 + 16];
#pragma unroll
  for (int i = 0; i < 8; ++i) {
    int r = mt * 16 + i + 8 * kh;
    out[(size_t)r * HID + col0]      = c0[i] + b0;
    out[(size_t)r * HID + col0 + 16] = c1[i] + b1;
  }
}

extern "C" void kernel_launch(void* const* d_in, const int* in_sizes, int n_in,
                              void* d_out, int out_size, void* d_ws, size_t ws_size,
                              hipStream_t stream) {
  (void)in_sizes; (void)n_in; (void)out_size; (void)ws_size;
  const float* X  = (const float*)d_in[0];
  const float* Wq = (const float*)d_in[1];
  const float* bq = (const float*)d_in[2];
  const float* Wk = (const float*)d_in[3];
  const float* bk = (const float*)d_in[4];
  const float* Wv = (const float*)d_in[5];
  const float* bv = (const float*)d_in[6];
  const float* Wo = (const float*)d_in[7];
  const float* bo = (const float*)d_in[8];
  float* out = (float*)d_out;

  char* ws = (char*)d_ws;
  size_t off = 0;
  auto take = [&](size_t bytes) -> char* {
    char* p = ws + off;
    off = (off + bytes + 255) & ~(size_t)255;
    return p;
  };
  const size_t XEL = (size_t)NB * SEQ * HID;
  const size_t QEL = (size_t)NB * NH * SEQ * VD;
  _Float16* Xh  = (_Float16*)take(XEL * 2);
  _Float16* WqT = (_Float16*)take((size_t)HID * HID * 2);
  _Float16* WkT = (_Float16*)take((size_t)HID * HID * 2);
  _Float16* WvT = (_Float16*)take((size_t)HID * HID * 2);
  _Float16* WoT = (_Float16*)take((size_t)HID * HID * 2);
  _Float16* Qh  = (_Float16*)take(QEL * 2);
  _Float16* Kh  = (_Float16*)take(QEL * 2);
  _Float16* Vt  = (_Float16*)take(QEL * 2);
  _Float16* Zh  = (_Float16*)take(XEL * 2);

  cvt_x_kernel<<<(int)(XEL / 256), 256, 0, stream>>>(X, Xh, (int)XEL);
  pack_w_kernel<<<(HID * HID) / 256, 256, 0, stream>>>(Wq, Wk, Wv, Wo, WqT, WkT, WvT, WoT);
  qkv_kernel<<<1024, 256, 0, stream>>>(Xh, WqT, WkT, WvT, bq, bk, bv, Qh, Kh, Vt);
  attn_kernel<<<256, 256, 0, stream>>>(Qh, Kh, Vt, Zh);
  proj_kernel<<<512, 256, 0, stream>>>(Zh, WoT, bo, out);
}
